// DeformationNetworkGraphConvolutionalLite_8504035246685
// MI455X (gfx1250) — matrix operand order repaired and checked by reading, compiled, not run on hardware
//
#include <hip/hip_runtime.h>

#define H_DIM 128
#define IMG_F 960

typedef __attribute__((ext_vector_type(16))) __bf16 v16bf;
typedef __attribute__((ext_vector_type(8)))  float  v8f;
typedef __attribute__((ext_vector_type(4)))  unsigned int v4u;
typedef __attribute__((ext_vector_type(8)))  int v8i_t;
typedef __attribute__((ext_vector_type(4)))  int v4i_t;

union FragBF { v16bf v; float4 f4[2]; };

__device__ __forceinline__ unsigned short bf_bits(float f) {
  union { float f; unsigned u; } c; c.f = f;
  unsigned u = c.u;
  return (unsigned short)((u + 0x7FFFu + ((u >> 16) & 1u)) >> 16); // RNE
}
__device__ __forceinline__ __bf16 f2bf(float f) {
  union { unsigned short s; __bf16 b; } c; c.s = bf_bits(f); return c.b;
}
__device__ __forceinline__ unsigned pack2bf(float a, float b) {
  return (unsigned)bf_bits(a) | ((unsigned)bf_bits(b) << 16);
}

// ---- WMMA fragment loaders (16-bit, per CDNA5 ISA 7.12.2 layouts) ----
__device__ __forceinline__ v16bf load_fragA(const __bf16* base, int stride, int kc, int lane) {
  int m  = lane & 15;
  int kb = (lane >> 4) << 3;
  const __bf16* p = base + (size_t)m * stride + kc + kb;
  FragBF fr;
  fr.f4[0] = *reinterpret_cast<const float4*>(p);
  fr.f4[1] = *reinterpret_cast<const float4*>(p + 16);
  return fr.v;
}
__device__ __forceinline__ v16bf load_fragB(const __bf16* W, int stride, int nBase, int kc, int lane) {
  int n  = nBase + (lane & 15);
  int kb = (lane >> 4) << 3;
  const __bf16* p = W + (size_t)n * stride + kc + kb;
  FragBF fr;
  fr.f4[0] = *reinterpret_cast<const float4*>(p);
  fr.f4[1] = *reinterpret_cast<const float4*>(p + 16);
  return fr.v;
}

// ---- TDM: async DMA of a contiguous run of bf16 elements into LDS ----
// 1-D tile (tile_dim0 elements); reads past tensor_dim0 return zero.
__device__ __forceinline__ void tdm_load_1d_bf16(const __bf16* gsrc, unsigned ldsOff,
                                                 unsigned tensorElems, unsigned tileElems) {
  unsigned long long ga = (unsigned long long)(uintptr_t)gsrc;
  v4u g0;
  g0.x = 1u;                                             // count=1 (valid descriptor)
  g0.y = ldsOff;                                         // LDS byte address
  g0.z = (unsigned)ga;                                   // global_addr[31:0]
  g0.w = (unsigned)((ga >> 32) & 0x01FFFFFFu) | (2u << 30); // global_addr[56:32] | type=2
  v8i_t g1;
  g1[0] = 0x00010000;                                    // data_size=1 (2 bytes/elem)
  g1[1] = (int)((tensorElems & 0xFFFFu) << 16);          // tensor_dim0[15:0]
  g1[2] = (int)((tensorElems >> 16) & 0xFFFFu);          // tensor_dim0[31:16]
  g1[3] = (int)((tileElems & 0xFFFFu) << 16);            // tile_dim0
  g1[4] = 0;                                             // tile_dim1=0, tile_dim2=0 (1-D)
  g1[5] = (int)H_DIM;                                    // tensor_dim0_stride lo32
  g1[6] = 0;
  g1[7] = 0;
  v4i_t g2 = {0, 0, 0, 0};
  v4i_t g3 = {0, 0, 0, 0};
#if __has_include(<hip/amd_detail/amd_gfx1250_TDM.h>)
  v8i_t g4 = {0, 0, 0, 0, 0, 0, 0, 0};
  __builtin_amdgcn_tensor_load_to_lds(g0, g1, g2, g3, g4, 0);   // clang-23 6-arg form
#else
  __builtin_amdgcn_tensor_load_to_lds(g0, g1, g2, g3, 0);       // ROCm 7.2 5-arg form
#endif
}

// ---------------- weight conversion ----------------
__global__ void cvt_f32_bf16(const float* __restrict__ src, __bf16* __restrict__ dst, int n) {
  int i = blockIdx.x * blockDim.x + threadIdx.x;
  if (i < n) dst[i] = f2bf(src[i]);
}
__global__ void cvt_layer0(const float* __restrict__ g0W0, const float* __restrict__ g0W1,
                           __bf16* __restrict__ W0b, __bf16* __restrict__ W1b) {
  int i = blockIdx.x * blockDim.x + threadIdx.x;
  if (i >= H_DIM * H_DIM) return;
  int r = i >> 7, k = i & 127;
  W0b[i] = f2bf(g0W0[r * 131 + k]);
  W1b[i] = f2bf(g0W1[r * 131 + k]);
}

// ---------------- backbone: x = relu(img @ bnW^T + b), bf16 out ----------------
// M=32 rows/block; 8 waves each own one N-tile of 16 and both M-tiles.
// (manual staging: needs fp32->bf16 conversion in flight, which TDM cannot do)
__global__ void __launch_bounds__(256)
backbone_gemm(const float* __restrict__ img, const __bf16* __restrict__ Wbn,
              const float* __restrict__ bnb, __bf16* __restrict__ xb, int Vtot) {
  __shared__ alignas(16) __bf16 tA[32 * IMG_F]; // 60 KB
  const int tid = threadIdx.x;
  const int rowBase = blockIdx.x * 32;
  for (int i4 = tid; i4 < 32 * IMG_F / 4; i4 += 256) {
    int i = i4 * 4;
    int row = rowBase + i / IMG_F;
    int col = i % IMG_F;
    if (row >= Vtot) row = Vtot - 1;
    float4 v = *reinterpret_cast<const float4*>(img + (size_t)row * IMG_F + col);
    uint2 o;
    o.x = pack2bf(v.x, v.y);
    o.y = pack2bf(v.z, v.w);
    reinterpret_cast<uint2*>(tA)[i4] = o;
  }
  __syncthreads();

  const int lane = tid & 31;
  const int w    = tid >> 5;
  v8f accLo = {}, accHi = {};
#pragma unroll 2
  for (int kc = 0; kc < IMG_F; kc += 32) {
    v16bf aLo = load_fragA(tA, IMG_F, kc, lane);
    v16bf aHi = load_fragA(tA + 16 * IMG_F, IMG_F, kc, lane);
    v16bf b   = load_fragB(Wbn, IMG_F, w * 16, kc, lane);
    accLo = __builtin_amdgcn_wmma_f32_16x16x32_bf16(false, aLo, false, b, (short)0, accLo, false, false);
    accHi = __builtin_amdgcn_wmma_f32_16x16x32_bf16(false, aHi, false, b, (short)0, accHi, false, false);
  }
  const int n = (lane & 15) + w * 16;
  const float bias = bnb[n];
  const int mBase = rowBase + ((lane >> 4) << 3);
  __bf16* p = xb + (size_t)mBase * H_DIM + n;
  if (rowBase + 32 <= Vtot) {
#pragma unroll
    for (int r = 0; r < 8; ++r) {
      p[(size_t)r * H_DIM]        = f2bf(fmaxf(accLo[r] + bias, 0.0f));
      p[(size_t)(r + 16) * H_DIM] = f2bf(fmaxf(accHi[r] + bias, 0.0f));
    }
  } else {
#pragma unroll
    for (int r = 0; r < 8; ++r) {
      if (mBase + r < Vtot)      p[(size_t)r * H_DIM]        = f2bf(fmaxf(accLo[r] + bias, 0.0f));
      if (mBase + r + 16 < Vtot) p[(size_t)(r + 16) * H_DIM] = f2bf(fmaxf(accHi[r] + bias, 0.0f));
    }
  }
}

// ---------------- graph-conv GEMM: y0 -> agg, y1 -> y1buf ----------------
// M=32 rows/block; A tile (32x128 bf16 = 8KB, contiguous) staged by the TDM.
__global__ void __launch_bounds__(256)
layer_gemm(const __bf16* __restrict__ xb, const __bf16* __restrict__ W0b,
           const __bf16* __restrict__ W1b, const float* __restrict__ b0,
           const float* __restrict__ b1, const float* __restrict__ verts,
           const float* __restrict__ W0e, const float* __restrict__ W1e,
           float* __restrict__ agg, float* __restrict__ y1, int Vtot) {
  __shared__ alignas(128) __bf16 tA[32 * H_DIM]; // 8 KB
  const int tid = threadIdx.x;
  const int rowBase = blockIdx.x * 32;
  if (tid < 32) { // wave 0 issues the tensor DMA; flat-LDS aperture => low 32 bits = LDS offset
    unsigned ldsOff = (unsigned)(uintptr_t)(void*)&tA[0];
    unsigned avail  = (unsigned)(Vtot - rowBase) * H_DIM;  // OOB rows read as zero
    tdm_load_1d_bf16(xb + (size_t)rowBase * H_DIM, ldsOff, avail, 32 * H_DIM);
    __builtin_amdgcn_s_wait_tensorcnt(0);
  }
  __syncthreads();

  const int lane = tid & 31;
  const int w    = tid >> 5;
  v8f acc0L = {}, acc0H = {}, acc1L = {}, acc1H = {};
#pragma unroll
  for (int kc = 0; kc < H_DIM; kc += 32) {
    v16bf aLo = load_fragA(tA, H_DIM, kc, lane);
    v16bf aHi = load_fragA(tA + 16 * H_DIM, H_DIM, kc, lane);
    v16bf f0  = load_fragB(W0b, H_DIM, w * 16, kc, lane);
    v16bf f1  = load_fragB(W1b, H_DIM, w * 16, kc, lane);
    acc0L = __builtin_amdgcn_wmma_f32_16x16x32_bf16(false, aLo, false, f0, (short)0, acc0L, false, false);
    acc0H = __builtin_amdgcn_wmma_f32_16x16x32_bf16(false, aHi, false, f0, (short)0, acc0H, false, false);
    acc1L = __builtin_amdgcn_wmma_f32_16x16x32_bf16(false, aLo, false, f1, (short)0, acc1L, false, false);
    acc1H = __builtin_amdgcn_wmma_f32_16x16x32_bf16(false, aHi, false, f1, (short)0, acc1H, false, false);
  }
  const int n = (lane & 15) + w * 16;
  const float bias0 = b0[n], bias1 = b1[n];
  float we0x = 0.f, we0y = 0.f, we0z = 0.f, we1x = 0.f, we1y = 0.f, we1z = 0.f;
  const bool hasV = (W0e != nullptr);
  if (hasV) { // rank-3 verts contribution of the 131-dim layer-0 input (fp32 exact)
    we0x = W0e[n * 131 + 128]; we0y = W0e[n * 131 + 129]; we0z = W0e[n * 131 + 130];
    we1x = W1e[n * 131 + 128]; we1y = W1e[n * 131 + 129]; we1z = W1e[n * 131 + 130];
  }
  const int mBase = rowBase + ((lane >> 4) << 3);
  float* pa = agg + (size_t)mBase * H_DIM + n;
  float* py = y1  + (size_t)mBase * H_DIM + n;
  const bool full = (rowBase + 32 <= Vtot);
#pragma unroll
  for (int half = 0; half < 2; ++half) {
    const v8f& c0 = half ? acc0H : acc0L;
    const v8f& c1 = half ? acc1H : acc1L;
    const int ro = half * 16;
#pragma unroll
    for (int r = 0; r < 8; ++r) {
      int m = mBase + ro + r;
      if (full || m < Vtot) {
        float v0 = c0[r] + bias0;
        float v1 = c1[r] + bias1;
        if (hasV) {
          float vx = verts[3 * m], vy = verts[3 * m + 1], vz = verts[3 * m + 2];
          v0 += vx * we0x + vy * we0y + vz * we0z;
          v1 += vx * we1x + vy * we1y + vz * we1z;
        }
        pa[(size_t)(ro + r) * H_DIM] = v0;
        py[(size_t)(ro + r) * H_DIM] = v1;
      }
    }
  }
}

// ---------------- edge scatter (undirected): agg[i]+=y1[j], agg[j]+=y1[i] ----------------
__global__ void __launch_bounds__(256)
edge_scatter(const int* __restrict__ edges, const float* __restrict__ y1,
             float* __restrict__ agg, int E) {
  int gid = blockIdx.x * blockDim.x + threadIdx.x;
  int e = gid >> 5;
  if (e >= E) return;
  int q = (gid & 31) << 2;
  int i = edges[2 * e], j = edges[2 * e + 1];
  float4 a = *reinterpret_cast<const float4*>(y1 + (size_t)j * H_DIM + q);
  float* pi = agg + (size_t)i * H_DIM + q;
  atomicAdd(pi + 0, a.x); atomicAdd(pi + 1, a.y);
  atomicAdd(pi + 2, a.z); atomicAdd(pi + 3, a.w);
  float4 b = *reinterpret_cast<const float4*>(y1 + (size_t)i * H_DIM + q);
  float* pj = agg + (size_t)j * H_DIM + q;
  atomicAdd(pj + 0, b.x); atomicAdd(pj + 1, b.y);
  atomicAdd(pj + 2, b.z); atomicAdd(pj + 3, b.w);
}

// ---------------- relu(agg) -> bf16 activations for next layer ----------------
__global__ void relu_to_bf16(const float* __restrict__ agg, __bf16* __restrict__ xb, int n4) {
  int i = blockIdx.x * blockDim.x + threadIdx.x;
  if (i >= n4) return;
  float4 v = reinterpret_cast<const float4*>(agg)[i];
  uint2 o;
  o.x = pack2bf(fmaxf(v.x, 0.f), fmaxf(v.y, 0.f));
  o.y = pack2bf(fmaxf(v.z, 0.f), fmaxf(v.w, 0.f));
  reinterpret_cast<uint2*>(xb)[i] = o;
}

// ---------------- output head: delta = relu(agg) @ voW^T + vob ----------------
__global__ void __launch_bounds__(256)
out_head(const float* __restrict__ agg, const float* __restrict__ voW,
         const float* __restrict__ vob, float* __restrict__ out, int Vtot) {
  int gid = blockIdx.x * blockDim.x + threadIdx.x;
  int v = gid >> 5, lane = gid & 31;
  if (v >= Vtot) return;
  int k = lane << 2;
  float4 xv = *reinterpret_cast<const float4*>(agg + (size_t)v * H_DIM + k);
  xv.x = fmaxf(xv.x, 0.f); xv.y = fmaxf(xv.y, 0.f);
  xv.z = fmaxf(xv.z, 0.f); xv.w = fmaxf(xv.w, 0.f);
  float s0 = 0.f, s1 = 0.f, s2 = 0.f;
#pragma unroll
  for (int t = 0; t < 4; ++t) {
    float xt = (&xv.x)[t];
    s0 += xt * voW[0 * H_DIM + k + t];
    s1 += xt * voW[1 * H_DIM + k + t];
    s2 += xt * voW[2 * H_DIM + k + t];
  }
#pragma unroll
  for (int off = 16; off > 0; off >>= 1) {
    s0 += __shfl_xor(s0, off, 32);
    s1 += __shfl_xor(s1, off, 32);
    s2 += __shfl_xor(s2, off, 32);
  }
  if (lane == 0) {
    out[(size_t)v * 3 + 0] = s0 + vob[0];
    out[(size_t)v * 3 + 1] = s1 + vob[1];
    out[(size_t)v * 3 + 2] = s2 + vob[2];
  }
}

extern "C" void kernel_launch(void* const* d_in, const int* in_sizes, int n_in,
                              void* d_out, int out_size, void* d_ws, size_t ws_size,
                              hipStream_t stream) {
  (void)n_in; (void)out_size; (void)ws_size;
  const float* img   = (const float*)d_in[0];
  const float* verts = (const float*)d_in[1];
  const int*   edges = (const int*)d_in[2];
  const float* bnW   = (const float*)d_in[3];
  const float* bnb   = (const float*)d_in[4];
  const float* g0W0  = (const float*)d_in[5];
  const float* g0b0  = (const float*)d_in[6];
  const float* g0W1  = (const float*)d_in[7];
  const float* g0b1  = (const float*)d_in[8];
  const float* gW0   = (const float*)d_in[9];
  const float* gb0   = (const float*)d_in[10];
  const float* gW1   = (const float*)d_in[11];
  const float* gb1   = (const float*)d_in[12];
  const float* voW   = (const float*)d_in[13];
  const float* vob   = (const float*)d_in[14];
  float* out = (float*)d_out;

  const int V = in_sizes[1] / 3;   // 200000
  const int E = in_sizes[2] / 2;   // 600000

  char* ws = (char*)d_ws;
  const size_t XB_BYTES  = (size_t)V * H_DIM * 2;
  const size_t F32_BYTES = (size_t)V * H_DIM * 4;
  __bf16* xb  = (__bf16*)(ws);
  float*  agg = (float*) (ws + XB_BYTES);
  float*  y1  = (float*) (ws + XB_BYTES + F32_BYTES);
  char*   wts = ws + XB_BYTES + 2 * F32_BYTES;
  __bf16* Wbn = (__bf16*)(wts);
  __bf16* W0b = (__bf16*)(wts + (size_t)H_DIM * IMG_F * 2);
  __bf16* W1b = W0b + 8 * H_DIM * H_DIM;

  cvt_f32_bf16<<<(H_DIM * IMG_F + 255) / 256, 256, 0, stream>>>(bnW, Wbn, H_DIM * IMG_F);
  cvt_layer0  <<<(H_DIM * H_DIM + 255) / 256, 256, 0, stream>>>(g0W0, g0W1, W0b, W1b);
  cvt_f32_bf16<<<(7 * H_DIM * H_DIM + 255) / 256, 256, 0, stream>>>(gW0, W0b + H_DIM * H_DIM, 7 * H_DIM * H_DIM);
  cvt_f32_bf16<<<(7 * H_DIM * H_DIM + 255) / 256, 256, 0, stream>>>(gW1, W1b + H_DIM * H_DIM, 7 * H_DIM * H_DIM);

  const int gemmBlocks = (V + 31) / 32;

  backbone_gemm<<<gemmBlocks, 256, 0, stream>>>(img, Wbn, bnb, xb, V);

  for (int l = 0; l < 8; ++l) {
    const __bf16* w0 = W0b + (size_t)l * H_DIM * H_DIM;
    const __bf16* w1 = W1b + (size_t)l * H_DIM * H_DIM;
    const float* bb0 = (l == 0) ? g0b0 : gb0 + (size_t)(l - 1) * H_DIM;
    const float* bb1 = (l == 0) ? g0b1 : gb1 + (size_t)(l - 1) * H_DIM;
    layer_gemm<<<gemmBlocks, 256, 0, stream>>>(
        xb, w0, w1, bb0, bb1,
        (l == 0) ? verts : nullptr,
        (l == 0) ? g0W0 : nullptr,
        (l == 0) ? g0W1 : nullptr,
        agg, y1, V);
    edge_scatter<<<((size_t)E * 32 + 255) / 256, 256, 0, stream>>>(edges, y1, agg, E);
    if (l < 7) {
      int n4 = V * H_DIM / 4;
      relu_to_bf16<<<(n4 + 255) / 256, 256, 0, stream>>>(agg, xb, n4);
    }
  }

  out_head<<<((size_t)V * 32 + 255) / 256, 256, 0, stream>>>(agg, voW, vob, out, V);
}
